// RingAttention_5025111736724
// MI455X (gfx1250) — matrix operand order repaired and checked by reading, not compiled, not run on hardware
//
#include <hip/hip_runtime.h>

// Flash attention forward, bf16 WMMA path for gfx1250 (MI455X).
// B=1, S=4096, Hq=16, Hkv=8 (GQA reps=2), D=128, fp32 in/out.
// Softmax in exp2 domain; B fragments group-double-buffered (8 frags /
// 16 ds_load_b128 in flight behind each batch of 8 WMMAs).

typedef __bf16 bf16_t;
typedef __attribute__((ext_vector_type(16))) __bf16 v16bf;
typedef __attribute__((ext_vector_type(8)))  __bf16 v8bf;
typedef __attribute__((ext_vector_type(4)))  __bf16 v4bf;
typedef __attribute__((ext_vector_type(8)))  float  v8f;

#define S_LEN 4096
#define HQ    16
#define HKV   8
#define DH    128
#define BQ    128          // queries per workgroup (8 waves x 16)
#define BK    128          // keys per block iteration
#define NKB   (S_LEN / BK)
#define NWAVES 8

static __device__ __forceinline__ bf16_t f2bf(float f) {
  return (bf16_t)f;   // native cvt, packs with neighbors
}

static __device__ __forceinline__ float redmax16(float x) {
  x = fmaxf(x, __shfl_xor(x, 1, 32));
  x = fmaxf(x, __shfl_xor(x, 2, 32));
  x = fmaxf(x, __shfl_xor(x, 4, 32));
  x = fmaxf(x, __shfl_xor(x, 8, 32));
  return x;
}

static __device__ __forceinline__ float redsum16(float x) {
  x += __shfl_xor(x, 1, 32);
  x += __shfl_xor(x, 2, 32);
  x += __shfl_xor(x, 4, 32);
  x += __shfl_xor(x, 8, 32);
  return x;
}

__global__ __launch_bounds__(256)
void fa_fwd_bf16wmma(const float* __restrict__ Qg, const float* __restrict__ Kg,
                     const float* __restrict__ Vg, float* __restrict__ Og) {
  extern __shared__ char smem_raw[];
  bf16_t* sK = (bf16_t*)smem_raw;      // [BK][DH]   row-major keys (bf16)
  bf16_t* sV = sK + BK * DH;           // [DH][BK]   transposed values (bf16)
  bf16_t* sP = sV + BK * DH;           // [NWAVES][16][BK] per-wave P scratch

  const int tid   = threadIdx.x;
  const int wave  = tid >> 5;
  const int lane  = tid & 31;
  const int l16   = lane & 15;
  const int lhalf = lane >> 4;         // 0: lanes 0-15, 1: lanes 16-31
  const int h     = blockIdx.y;        // query head
  const int hk    = h >> 1;            // GQA: kv head = h / reps, reps = 2
  const int qbase = blockIdx.x * BQ + wave * 16;

  // 1/sqrt(128) * log2(e): scores come out in the exp2 domain
  const float scale = 0.08838834764831845f * 1.4426950408889634f;

  // ---- Q A-fragments (16x32 bf16 per WMMA, 4 chunks cover D=128) ----
  v16bf qa[4];
  {
    const float* qrow = Qg + ((long)(qbase + l16) * HQ + h) * DH;
#pragma unroll
    for (int c = 0; c < 4; ++c) {
      const int d0 = c * 32 + lhalf * 16;
#pragma unroll
      for (int j = 0; j < 8; ++j) {
        qa[c][j]     = f2bf(qrow[d0 + j] * scale);
        qa[c][8 + j] = f2bf(qrow[d0 + 8 + j] * scale);
      }
    }
  }

  // ---- running softmax state + O accumulators (C layout: row = r + 8*lhalf) ----
  float m_run[8], l_run[8];
  v8f oacc[8];
  const v8f vzero = {0.f, 0.f, 0.f, 0.f, 0.f, 0.f, 0.f, 0.f};
#pragma unroll
  for (int r = 0; r < 8; ++r) { m_run[r] = -__builtin_inff(); l_run[r] = 0.f; }
#pragma unroll
  for (int dt = 0; dt < 8; ++dt) oacc[dt] = vzero;

  bf16_t* sPw = sP + wave * 16 * BK;

  // ---- B-fragment group helpers (8 frags = one K-chunk, all tiles) ----
  auto ldgK = [&](v16bf* buf, int c) {
#pragma unroll
    for (int nt = 0; nt < 8; ++nt)
      buf[nt] = *(const v16bf*)&sK[(nt * 16 + l16) * DH + c * 32 + lhalf * 16];
  };
  auto ldgV = [&](v16bf* buf, int c) {
#pragma unroll
    for (int dt = 0; dt < 8; ++dt)
      buf[dt] = *(const v16bf*)&sV[(dt * 16 + l16) * BK + c * 32 + lhalf * 16];
  };
  auto mmag = [&](v8f* acc, const v16bf a, const v16bf* buf) {
#pragma unroll
    for (int t = 0; t < 8; ++t)
      acc[t] = __builtin_amdgcn_wmma_f32_16x16x32_bf16(
          false, a, false, buf[t], (short)0, acc[t], false, false);
  };

#pragma unroll 1
  for (int kb = 0; kb < NKB; ++kb) {
    __syncthreads();  // previous iteration's LDS reads done

    // ---- stage K block -> LDS, row-major bf16 ----
    {
      const int r = tid >> 1;
      const int colbase = (tid & 1) * 64;
      const float* krow = Kg + ((long)(kb * BK + r) * HKV + hk) * DH + colbase;
      if (kb + 1 < NKB)
        __builtin_prefetch(Kg + ((long)((kb + 1) * BK + r) * HKV + hk) * DH + colbase, 0, 0);
#pragma unroll
      for (int g = 0; g < 8; ++g) {
        float4 x0 = ((const float4*)krow)[2 * g];
        float4 x1 = ((const float4*)krow)[2 * g + 1];
        v8bf o;
        o[0] = f2bf(x0.x); o[1] = f2bf(x0.y); o[2] = f2bf(x0.z); o[3] = f2bf(x0.w);
        o[4] = f2bf(x1.x); o[5] = f2bf(x1.y); o[6] = f2bf(x1.z); o[7] = f2bf(x1.w);
        *(v8bf*)&sK[r * DH + colbase + g * 8] = o;
      }
    }

    // ---- stage V block -> LDS transposed [d][key], bf16 ----
#pragma unroll
    for (int i = 0; i < 2; ++i) {
      const int st   = tid + 256 * i;       // 512 subtiles of 4 keys x 8 d
      const int key0 = (st & 31) * 4;
      const int d0   = (st >> 5) * 8;
      float av[4][8];
#pragma unroll
      for (int kk = 0; kk < 4; ++kk) {
        const float* vrow = Vg + ((long)(kb * BK + key0 + kk) * HKV + hk) * DH + d0;
        float4 x0 = ((const float4*)vrow)[0];
        float4 x1 = ((const float4*)vrow)[1];
        av[kk][0] = x0.x; av[kk][1] = x0.y; av[kk][2] = x0.z; av[kk][3] = x0.w;
        av[kk][4] = x1.x; av[kk][5] = x1.y; av[kk][6] = x1.z; av[kk][7] = x1.w;
      }
#pragma unroll
      for (int dd = 0; dd < 8; ++dd) {
        v4bf p;
        p[0] = f2bf(av[0][dd]); p[1] = f2bf(av[1][dd]);
        p[2] = f2bf(av[2][dd]); p[3] = f2bf(av[3][dd]);
        *(v4bf*)&sV[(d0 + dd) * BK + key0] = p;
      }
    }
    __syncthreads();

    // ---- scores S = (Q*scale) K^T : group double-buffered over 4 chunks ----
    v8f sacc[8];
#pragma unroll
    for (int nt = 0; nt < 8; ++nt) sacc[nt] = vzero;
    {
      v16bf bufA[8], bufB[8];
      ldgK(bufA, 0);
      ldgK(bufB, 1);          // 16 loads outstanding behind chunk-0 WMMAs
      mmag(sacc, qa[0], bufA);
      ldgK(bufA, 2);
      mmag(sacc, qa[1], bufB);
      ldgK(bufB, 3);
      mmag(sacc, qa[2], bufA);
      mmag(sacc, qa[3], bufB);
    }

    // ---- online softmax (exp2 domain) over this key block ----
    float mnew[8], alpha[8], lsum[8];
#pragma unroll
    for (int r = 0; r < 8; ++r) {
      float mx = sacc[0][r];
#pragma unroll
      for (int nt = 1; nt < 8; ++nt) mx = fmaxf(mx, sacc[nt][r]);
      mx = redmax16(mx);
      mnew[r]  = fmaxf(m_run[r], mx);
      alpha[r] = exp2f(m_run[r] - mnew[r]);   // 0 on first block
      m_run[r] = mnew[r];
      lsum[r]  = 0.f;
    }
#pragma unroll
    for (int nt = 0; nt < 8; ++nt) {
#pragma unroll
      for (int r = 0; r < 8; ++r) {
        const float p = exp2f(sacc[nt][r] - mnew[r]);
        sacc[nt][r] = p;
        lsum[r] += p;
      }
    }
#pragma unroll
    for (int r = 0; r < 8; ++r) {
      lsum[r]  = redsum16(lsum[r]);
      l_run[r] = l_run[r] * alpha[r] + lsum[r];
#pragma unroll
      for (int dt = 0; dt < 8; ++dt) oacc[dt][r] *= alpha[r];
    }

    // ---- re-layout P (C layout -> A layout) via wave-private LDS ----
    // LDS ops are in-order within a wave; no barrier needed.
#pragma unroll
    for (int nt = 0; nt < 8; ++nt) {
#pragma unroll
      for (int r = 0; r < 8; ++r)
        sPw[(r + 8 * lhalf) * BK + nt * 16 + l16] = f2bf(sacc[nt][r]);
    }

    v16bf pa[4];
#pragma unroll
    for (int c = 0; c < 4; ++c) {
      const bf16_t* prow = sPw + l16 * BK + c * 32 + lhalf * 16;
      v8bf lo = *(const v8bf*)prow;
      v8bf hi = *(const v8bf*)(prow + 8);
#pragma unroll
      for (int j = 0; j < 8; ++j) { pa[c][j] = lo[j]; pa[c][8 + j] = hi[j]; }
    }

    // ---- O += P V : group double-buffered over 4 key chunks ----
    {
      v16bf bufA[8], bufB[8];
      ldgV(bufA, 0);
      ldgV(bufB, 1);
      mmag(oacc, pa[0], bufA);
      ldgV(bufA, 2);
      mmag(oacc, pa[1], bufB);
      ldgV(bufB, 3);
      mmag(oacc, pa[2], bufA);
      mmag(oacc, pa[3], bufB);
    }
  }

  // ---- epilogue: O / l, coalesced stores (lanes 0-15 cover consecutive d) ----
#pragma unroll
  for (int r = 0; r < 8; ++r) {
    const float inv = 1.0f / l_run[r];
    const int row = qbase + r + 8 * lhalf;
    float* orow = Og + ((long)row * HQ + h) * DH;
#pragma unroll
    for (int dt = 0; dt < 8; ++dt)
      orow[dt * 16 + l16] = oacc[dt][r] * inv;
  }
}

extern "C" void kernel_launch(void* const* d_in, const int* in_sizes, int n_in,
                              void* d_out, int out_size, void* d_ws, size_t ws_size,
                              hipStream_t stream) {
  (void)in_sizes; (void)n_in; (void)out_size; (void)d_ws; (void)ws_size;
  const float* Q = (const float*)d_in[0];
  const float* K = (const float*)d_in[1];
  const float* V = (const float*)d_in[2];
  float* O = (float*)d_out;

  const size_t smem = (size_t)(BK * DH + BK * DH + NWAVES * 16 * BK) * 2u;  // 96 KB
  dim3 grid(S_LEN / BQ, HQ);  // 32 x 16 workgroups
  fa_fwd_bf16wmma<<<grid, 256, smem, stream>>>(Q, K, V, O);
}
